// BYOSv1_61211873903141
// MI455X (gfx1250) — matrix operand order repaired and checked by reading, not compiled, not run on hardware
//
#include <hip/hip_runtime.h>
#include <hip/hip_bf16.h>

typedef __bf16 bf16_t;
typedef __attribute__((ext_vector_type(16))) __bf16 v16bf;
typedef __attribute__((ext_vector_type(8)))  __bf16 v8bf;
typedef __attribute__((ext_vector_type(8)))  float  v8f;
typedef __attribute__((ext_vector_type(4)))  unsigned int v4u;
typedef __attribute__((ext_vector_type(8)))  int v8i;
typedef __attribute__((ext_vector_type(4)))  int v4i;

#define B_    4
#define T_    4096
#define D_    1024
#define NH_   16
#define NS_   64
#define HD_   64
#define TC_   64                 // time-chunk length
#define NC_   (T_ / TC_)         // 64 chunks per (b,h)
#define KT_   128                // k-block width staged per TDM transfer
#define NKB_  (D_ / KT_)         // 8 k-blocks
#define ALPHA_ 0.1f

// workspace layout (bytes)
#define OFF_HB  0ull                       // bf16 H          : 32 MB
#define OFF_KP  33554432ull                // bf16 KP[h,s,i]  :  2 MB
#define OFF_C   35651584ull                // f32 C[bh,c,s,d] : 64 MB
#define OFF_G   102760448ull               // f32 G[bh,c,s]   :  1 MB

#if defined(__has_builtin)
#if __has_builtin(__builtin_amdgcn_tensor_load_to_lds)
#define HAVE_TDM 1
#endif
#endif

// ---------------------------------------------------------------------------
// TDM descriptor: load a (rows=64) x (cols=KT_) bf16 tile from a row-major
// tensor with row stride D_ elements into LDS at lds_addr. Packed per CDNA5
// ISA 8.3/8.4: group0 = {count|flags, lds_addr, global_addr lo/hi|type=2},
// group1 = {mask/data_size, dims, tile dims, strides}.
// ---------------------------------------------------------------------------
#if HAVE_TDM
__device__ __forceinline__ void tdm_load_tile(unsigned lds_addr, const bf16_t* gptr) {
    unsigned long long ga = (unsigned long long)(uintptr_t)gptr;
    v4u g0;
    g0[0] = 1u;                                              // count=1 (valid user D#)
    g0[1] = lds_addr;                                        // LDS byte address
    g0[2] = (unsigned)(ga & 0xFFFFFFFFu);                    // global_addr[31:0]
    g0[3] = (unsigned)((ga >> 32) & 0x01FFFFFFu) | (2u << 30); // [24:0]=addr[56:32], type=2
    v8i g1;
    g1[0] = 0x00010000;                                      // data_size=1 (2 bytes)
    g1[1] = (int)(((unsigned)D_  & 0xFFFFu) << 16);          // tensor_dim0 lo16 (=1024)
    g1[2] = (int)(((unsigned)TC_ & 0xFFFFu) << 16);          // dim0 hi=0 | tensor_dim1 lo16 (=64)
    g1[3] = (int)(((unsigned)KT_ & 0xFFFFu) << 16);          // dim1 hi=0 | tile_dim0 (=128)
    g1[4] = TC_;                                             // tile_dim1 (=64) | tile_dim2=0
    g1[5] = D_;                                              // tensor_dim0_stride lo32 (=1024)
    g1[6] = 0;                                               // stride hi | dim1_stride lo
    g1[7] = 0;
    v4i z4 = {0, 0, 0, 0};
#if __clang_major__ >= 23
    v8i z8 = {0, 0, 0, 0, 0, 0, 0, 0};
    __builtin_amdgcn_tensor_load_to_lds(g0, g1, z4, z4, z8, 0);
#else
    __builtin_amdgcn_tensor_load_to_lds(g0, g1, z4, z4, 0);
#endif
}
#endif

// ---------------------------------------------------------------------------
// K1: convert H (f32) -> bf16, 4 elements/thread
// ---------------------------------------------------------------------------
__global__ __launch_bounds__(256) void k_convH(const float* __restrict__ H,
                                               bf16_t* __restrict__ Hb) {
    int idx = (blockIdx.x * 256 + threadIdx.x) * 4;
    float4 v = *(const float4*)(H + idx);
    bf16_t o[4] = {(bf16_t)v.x, (bf16_t)v.y, (bf16_t)v.z, (bf16_t)v.w};
    *(uint2*)(Hb + idx) = *(const uint2*)o;
}

// ---------------------------------------------------------------------------
// K2: KP[h,s,i] = 0.125 * sum_d W[h*64+d, i] * proto[s, h*64+d]  (bf16 out)
// ---------------------------------------------------------------------------
__global__ __launch_bounds__(256) void k_kp(const float* __restrict__ W,
                                            const float* __restrict__ proto,
                                            bf16_t* __restrict__ KP) {
    int h = blockIdx.x >> 6;
    int s = blockIdx.x & 63;
    int tid = threadIdx.x;
    float acc[4] = {0.f, 0.f, 0.f, 0.f};
    for (int d = 0; d < 64; ++d) {
        float p = proto[(size_t)s * D_ + h * 64 + d];
        const float* wr = W + (size_t)(h * 64 + d) * D_;
#pragma unroll
        for (int j = 0; j < 4; ++j) acc[j] += p * wr[tid + 256 * j];
    }
#pragma unroll
    for (int j = 0; j < 4; ++j)
        KP[(size_t)(h * 64 + s) * D_ + tid + 256 * j] = (bf16_t)(0.125f * acc[j]);
}

// ---------------------------------------------------------------------------
// K3: per (b,h,chunk) workgroup, 256 threads = 8 waves.
//   score WMMA (A staged by TDM into LDS double-buffer, B from global)
//   -> softmax -> per-slot backward scan -> contribution WMMA.
// ---------------------------------------------------------------------------
__global__ __launch_bounds__(256) void k_chunk(const bf16_t* __restrict__ Hb,
                                               const bf16_t* __restrict__ KP,
                                               float* __restrict__ Cws,
                                               float* __restrict__ Gws) {
    int blk  = blockIdx.x;            // ((b*NH + h)*NC + c)
    int c    = blk & (NC_ - 1);
    int h    = (blk >> 6) & (NH_ - 1);
    int b    = blk >> 10;
    int tid  = threadIdx.x;
    int lane = tid & 31;
    int wave = tid >> 5;
    int lrow = lane & 15;
    int lhi  = lane >> 4;

    __shared__ float sS[64][65];                       // scores -> softmax w
    __shared__ __align__(32) bf16_t weffT[64][80];     // [s][t]
    __shared__ __align__(32) bf16_t Bm[64][80];        // [d][t] head slice, transposed
    __shared__ __align__(32) bf16_t Abuf[2][TC_][KT_]; // TDM-staged A k-slices

    const int t0 = c * TC_;
    const bf16_t* Hrow0 = Hb + (size_t)(b * T_ + t0) * D_;

    // zero-fill A buffer once (also guarantees visible LDS stores for the
    // compiler; the DMA overwrites this).
    {
        v8bf z = {};
        bf16_t* base = &Abuf[0][0][0] + tid * (2 * TC_ * KT_ / 256);
#pragma unroll
        for (int j = 0; j < (2 * TC_ * KT_ / 256) / 8; ++j)
            *(v8bf*)(base + 8 * j) = z;
    }

    // stage Bm[d][t] = Hb[b, t0+t, h*64+d] (transposed head slice)
    {
        int t  = tid >> 2;
        int dg = (tid & 3) * 16;
        const bf16_t* src = Hrow0 + (size_t)t * D_ + h * 64 + dg;
#pragma unroll
        for (int j = 0; j < 16; ++j) Bm[dg + j][t] = src[j];
    }
    __syncthreads();

#if HAVE_TDM
    unsigned abuf_addr[2];
    abuf_addr[0] = (unsigned)(uintptr_t)&Abuf[0][0][0];
    abuf_addr[1] = (unsigned)(uintptr_t)&Abuf[1][0][0];
    if (wave == 0) {
        tdm_load_tile(abuf_addr[0], Hrow0);              // prologue: k-block 0
        __builtin_amdgcn_s_wait_tensorcnt(0);
    }
    __syncthreads();
#else
    // fallback: manual staging of k-block 0
    {
        int t = tid >> 2;
        int k0 = (tid & 3) * 32;
        const bf16_t* src = Hrow0 + (size_t)t * D_ + k0;
#pragma unroll
        for (int j = 0; j < 4; ++j)
            *(v8bf*)(&Abuf[0][t][k0 + 8 * j]) = *(const v8bf*)(src + 8 * j);
    }
    __syncthreads();
#endif

    // ---- score GEMM: 4x4 grid of 16x16 tiles, 2 tiles per wave ----
    int tm  = wave & 3;               // t-tile row
    int tn0 = (wave >> 2) * 2;        // first s-tile col
    v8f acc0 = {}; v8f acc1 = {};
    const bf16_t* bp0 = KP + (size_t)(h * 64 + tn0 * 16 + lrow) * D_ + lhi * 16;
    const bf16_t* bp1 = bp0 + (size_t)16 * D_;

    for (int kb = 0; kb < NKB_; ++kb) {
#if HAVE_TDM
        if (kb + 1 < NKB_ && wave == 0)
            tdm_load_tile(abuf_addr[(kb + 1) & 1], Hrow0 + (kb + 1) * KT_);
#endif
        const bf16_t* ap = &Abuf[kb & 1][tm * 16 + lrow][lhi * 16];
        const bf16_t* bq0 = bp0 + kb * KT_;
        const bf16_t* bq1 = bp1 + kb * KT_;
#pragma unroll
        for (int ks = 0; ks < KT_ / 32; ++ks) {
            int k = ks * 32;
            v8bf alo = *(const v8bf*)(ap + k);
            v8bf ahi = *(const v8bf*)(ap + k + 8);
            v16bf A;
#pragma unroll
            for (int j = 0; j < 8; ++j) { A[j] = alo[j]; A[j + 8] = ahi[j]; }
            v16bf B0 = *(const v16bf*)(bq0 + k);
            v16bf B1 = *(const v16bf*)(bq1 + k);
            acc0 = __builtin_amdgcn_wmma_f32_16x16x32_bf16(false, A, false, B0, (short)0, acc0, false, false);
            acc1 = __builtin_amdgcn_wmma_f32_16x16x32_bf16(false, A, false, B1, (short)0, acc1, false, false);
        }
#if HAVE_TDM
        if (kb + 1 < NKB_ && wave == 0)
            __builtin_amdgcn_s_wait_tensorcnt(0);        // next buffer landed
        __syncthreads();
#else
        __syncthreads();
        if (kb + 1 < NKB_) {
            int t = tid >> 2;
            int k0 = (tid & 3) * 32;
            const bf16_t* src = Hrow0 + (size_t)t * D_ + (kb + 1) * KT_ + k0;
#pragma unroll
            for (int j = 0; j < 4; ++j)
                *(v8bf*)(&Abuf[(kb + 1) & 1][t][k0 + 8 * j]) = *(const v8bf*)(src + 8 * j);
        }
        __syncthreads();
#endif
    }
#pragma unroll
    for (int r = 0; r < 8; ++r) {
        int m = tm * 16 + r + lhi * 8;
        sS[m][tn0 * 16 + lrow]       = acc0[r];
        sS[m][(tn0 + 1) * 16 + lrow] = acc1[r];
    }
    __syncthreads();

    // ---- softmax over s (64 slots) per row t ----
    if (tid < 64) {
        int t = tid;
        float mx = -1e30f;
        for (int s = 0; s < 64; ++s) mx = fmaxf(mx, sS[t][s]);
        float sum = 0.f;
        for (int s = 0; s < 64; ++s) { float e = __expf(sS[t][s] - mx); sS[t][s] = e; sum += e; }
        float inv = 1.f / sum;
        for (int s = 0; s < 64; ++s) sS[t][s] *= inv;
    }
    __syncthreads();

    // ---- backward scan over t per slot s: local exclusive suffix products ----
    if (tid < 64) {
        int s = tid;
        float P = 1.f;
        for (int t = TC_ - 1; t >= 0; --t) {
            float w = sS[t][s];
            weffT[s][t] = (bf16_t)(ALPHA_ * w * P);
            P *= (1.f - ALPHA_ * w);
        }
        Gws[(size_t)blk * 64 + s] = P;   // chunk gate product per slot
    }
    __syncthreads();

    // ---- contribution GEMM: C[s][d] = sum_t weffT[s][t] * Bm[d][t] ----
    v8f c0 = {}; v8f c1 = {};
    const bf16_t* a2  = &weffT[tm * 16 + lrow][lhi * 16];
    const bf16_t* bb0 = &Bm[tn0 * 16 + lrow][lhi * 16];
    const bf16_t* bb1 = &Bm[(tn0 + 1) * 16 + lrow][lhi * 16];
#pragma unroll
    for (int k = 0; k < TC_; k += 32) {
        v8bf alo = *(const v8bf*)(a2 + k);
        v8bf ahi = *(const v8bf*)(a2 + k + 8);
        v16bf A;
#pragma unroll
        for (int j = 0; j < 8; ++j) { A[j] = alo[j]; A[j + 8] = ahi[j]; }
        v16bf B0 = *(const v16bf*)(bb0 + k);
        v16bf B1 = *(const v16bf*)(bb1 + k);
        c0 = __builtin_amdgcn_wmma_f32_16x16x32_bf16(false, A, false, B0, (short)0, c0, false, false);
        c1 = __builtin_amdgcn_wmma_f32_16x16x32_bf16(false, A, false, B1, (short)0, c1, false, false);
    }
    float* Cout = Cws + (size_t)blk * (64 * 64);
#pragma unroll
    for (int r = 0; r < 8; ++r) {
        int m = tm * 16 + r + lhi * 8;               // s index
        Cout[m * 64 + tn0 * 16 + lrow]       = c0[r];
        Cout[m * 64 + (tn0 + 1) * 16 + lrow] = c1[r];
    }
}

// ---------------------------------------------------------------------------
// K4: cross-chunk suffix combine. One block per (b,h); thread owns one s and
// 16 d values. acc += P*C_c ; P *= G_c, scanning chunks last->first.
// ---------------------------------------------------------------------------
__global__ __launch_bounds__(256) void k_combine(const float* __restrict__ Cws,
                                                 const float* __restrict__ Gws,
                                                 float* __restrict__ out) {
    int bh  = blockIdx.x;             // b*NH + h
    int b   = bh >> 4;
    int h   = bh & 15;
    int tid = threadIdx.x;
    int s   = tid >> 2;
    int d0  = (tid & 3) * 16;
    float acc[16];
#pragma unroll
    for (int j = 0; j < 16; ++j) acc[j] = 0.f;
    float P = 1.f;
    for (int c = NC_ - 1; c >= 0; --c) {
        size_t base = (size_t)bh * NC_ + c;
        const float* Cp = Cws + base * (64 * 64) + s * 64 + d0;
#pragma unroll
        for (int j = 0; j < 16; ++j) acc[j] += P * Cp[j];
        P *= Gws[base * 64 + s];
    }
    float* op = out + (size_t)(b * 64 + s) * D_ + h * 64 + d0;
#pragma unroll
    for (int j = 0; j < 16; ++j) op[j] = acc[j];
}

// ---------------------------------------------------------------------------
extern "C" void kernel_launch(void* const* d_in, const int* in_sizes, int n_in,
                              void* d_out, int out_size, void* d_ws, size_t ws_size,
                              hipStream_t stream) {
    const float* H     = (const float*)d_in[0];   // [B,T,D]
    const float* proto = (const float*)d_in[1];   // [NS,D]
    const float* W     = (const float*)d_in[2];   // [D,D]
    float* out = (float*)d_out;                   // [B,NS,D]

    char* ws = (char*)d_ws;
    bf16_t* Hb  = (bf16_t*)(ws + OFF_HB);
    bf16_t* KP  = (bf16_t*)(ws + OFF_KP);
    float*  Cws = (float*)(ws + OFF_C);
    float*  Gws = (float*)(ws + OFF_G);

    k_convH<<<(B_ * T_ * D_) / (256 * 4), 256, 0, stream>>>(H, Hb);
    k_kp<<<NH_ * NS_, 256, 0, stream>>>(W, proto, KP);
    k_chunk<<<B_ * NH_ * NC_, 256, 0, stream>>>(Hb, KP, Cws, Gws);
    k_combine<<<B_ * NH_, 256, 0, stream>>>(Cws, Gws, out);
}
